// HSTUForSequenceModeling_4587025072282
// MI455X (gfx1250) — matrix operand order repaired and checked by reading, not compiled, not run on hardware
//
#include <hip/hip_runtime.h>

// ---------------------------------------------------------------------------
// HSTU-style transformer forward for MI455X (gfx1250), bf16 WMMA GEMM path.
// B=8, S=512, D=1024, H=16, DH=64, L=6, V=32000
// ---------------------------------------------------------------------------

#define Bn 8
#define Sn 512
#define Dn 1024
#define Hn 16
#define DHn 64
#define Ln 6
#define Vn 32000
#define MROWS (Bn * Sn)   // 4096

typedef __attribute__((ext_vector_type(16))) __bf16 v16bf;
typedef __attribute__((ext_vector_type(2)))  __bf16 v2bf;
typedef __attribute__((ext_vector_type(8)))  float  v8f;

union U16 { unsigned u[8]; v16bf v; };

static __device__ inline unsigned short f2bf(float f) {
    unsigned x = __float_as_uint(f);
    x += 0x7FFFu + ((x >> 16) & 1u);     // round to nearest even
    return (unsigned short)(x >> 16);
}
static __device__ inline unsigned pack2(float lo, float hi) {
#if __has_builtin(__builtin_amdgcn_cvt_pk_bf16_f32)
    v2bf p = __builtin_amdgcn_cvt_pk_bf16_f32(lo, hi);
    return __builtin_bit_cast(unsigned, p);
#else
    return (unsigned)f2bf(lo) | ((unsigned)f2bf(hi) << 16);
#endif
}
static __device__ inline float fast_sigmoid(float x) {
    return __builtin_amdgcn_rcpf(1.f + __expf(-x));
}

static __device__ inline v8f wmma_bf16(const U16& a, const U16& b, v8f c) {
    return __builtin_amdgcn_wmma_f32_16x16x32_bf16(
        /*neg_a=*/false, a.v, /*neg_b=*/false, b.v,
        /*c_mod=*/(short)0, c, /*reuse_a=*/false, /*reuse_b=*/false);
}

// ---------------------------------------------------------------------------
// 1) Embedding gather + positional encoding
// ---------------------------------------------------------------------------
__global__ __launch_bounds__(256)
void embed_kernel(const int* __restrict__ ids, const float* __restrict__ emb,
                  const float* __restrict__ pos, float* __restrict__ x)
{
    const int row = blockIdx.x;           // 0..4095
    const int s   = row & (Sn - 1);
    const int t   = threadIdx.x;
    const long id = ids[row];
#pragma unroll
    for (int i = 0; i < Dn / 256; ++i) {
        int c = t + 256 * i;
        x[(size_t)row * Dn + c] = emb[(size_t)id * Dn + c] + pos[(size_t)s * Dn + c];
    }
}

// ---------------------------------------------------------------------------
// 2) LayerNorm (one row per block): wave32 butterfly + 8-partial LDS exchange
// ---------------------------------------------------------------------------
__global__ __launch_bounds__(256)
void ln_kernel(const float* __restrict__ x, const float* __restrict__ g,
               const float* __restrict__ b, float* __restrict__ xn)
{
    const int row = blockIdx.x;
    const int t   = threadIdx.x;
    const int w   = t >> 5, lane = t & 31;
    const float* xr = x + (size_t)row * Dn;
    float v[Dn / 256];
    float s = 0.f, s2 = 0.f;
#pragma unroll
    for (int i = 0; i < Dn / 256; ++i) {
        v[i] = xr[t + 256 * i];
        s += v[i];
        s2 += v[i] * v[i];
    }
    // intra-wave butterfly reduction
#pragma unroll
    for (int o = 16; o > 0; o >>= 1) {
        s  += __shfl_xor(s,  o, 32);
        s2 += __shfl_xor(s2, o, 32);
    }
    __shared__ float ps[8], ps2[8];
    if (lane == 0) { ps[w] = s; ps2[w] = s2; }
    __syncthreads();
    float ts = 0.f, ts2 = 0.f;
#pragma unroll
    for (int i = 0; i < 8; ++i) { ts += ps[i]; ts2 += ps2[i]; }
    const float mu  = ts * (1.f / Dn);
    const float var = ts2 * (1.f / Dn) - mu * mu;
    const float inv = rsqrtf(var + 1e-5f);
#pragma unroll
    for (int i = 0; i < Dn / 256; ++i) {
        int c = t + 256 * i;
        xn[(size_t)row * Dn + c] = (v[i] - mu) * inv * g[c] + b[c];
    }
}

// ---------------------------------------------------------------------------
// 3) bf16 WMMA GEMM:  C = act(A[M,K] @ W[K,N] + bias) (+resid)
//    128x128 tile per block, BK=64, 256 threads = 8 waves (4m x 2n).
//    Each wave: 32x64 sub-tile = 2x4 C fragments; 16 v_wmma per k-stage.
//    Software pipeline: next k-stage's fp32 tiles are loaded (as float2) and
//    pair-packed to bf16 registers while the current stage's WMMAs run.
// ---------------------------------------------------------------------------
template<int ACT, bool HAS_RESID, bool NT_STORE>
__global__ __launch_bounds__(256)
void gemm_bf16_kernel(const float* __restrict__ A, const float* __restrict__ W,
                      const float* __restrict__ bias, const float* __restrict__ resid,
                      float* __restrict__ C, int M, int N, int K)
{
    const int t    = threadIdx.x;
    const int lane = t & 31;
    const int w    = t >> 5;
    const int wm   = w & 3;          // 4 m-slots of 32 rows
    const int wn   = w >> 2;         // 2 n-slots of 64 cols
    const int half = lane >> 4, l16 = lane & 15;
    const int n0 = blockIdx.x * 128;
    const int m0 = blockIdx.y * 128;

    __shared__ unsigned short As[128][66];   // [m][k]
    __shared__ unsigned short Bt[128][66];   // [n][k]  (W transposed)

    const v8f zero = {0.f, 0.f, 0.f, 0.f, 0.f, 0.f, 0.f, 0.f};
    v8f acc[2][4];
#pragma unroll
    for (int fa = 0; fa < 2; ++fa)
#pragma unroll
        for (int fb = 0; fb < 4; ++fb) acc[fa][fb] = zero;

    // ---- prefetch registers: 16 packed-pair uints per tile ----
    unsigned pA[16], pW[16];

    auto loadA = [&](int k0) {
#pragma unroll
        for (int p = 0; p < 16; ++p) {
            int pi = p * 256 + t;
            int r = pi >> 5, c = (pi & 31) * 2;
            const float2 f = *(const float2*)&A[(size_t)(m0 + r) * K + k0 + c];
            pA[p] = pack2(f.x, f.y);
        }
    };
    auto loadW = [&](int k0) {
#pragma unroll
        for (int p = 0; p < 16; ++p) {
            int pi = p * 256 + t;
            int kk = pi >> 6, n = (pi & 63) * 2;
            const float2 f = *(const float2*)&W[(size_t)(k0 + kk) * N + n0 + n];
            pW[p] = pack2(f.x, f.y);
        }
    };

    loadA(0);
    loadW(0);

    for (int k0 = 0; k0 < K; k0 += 64) {
        // store prefetched tile to LDS
#pragma unroll
        for (int p = 0; p < 16; ++p) {
            int pi = p * 256 + t;
            int r = pi >> 5, c = (pi & 31) * 2;
            *(unsigned*)&As[r][c] = pA[p];
        }
#pragma unroll
        for (int p = 0; p < 16; ++p) {
            int pi = p * 256 + t;
            int kk = pi >> 6, n = (pi & 63) * 2;
            unsigned v = pW[p];
            Bt[n][kk]     = (unsigned short)(v & 0xFFFFu);
            Bt[n + 1][kk] = (unsigned short)(v >> 16);
        }
        __syncthreads();

        // kick off next stage's global loads (overlap with WMMA below)
        if (k0 + 64 < K) { loadA(k0 + 64); loadW(k0 + 64); }

#pragma unroll
        for (int s = 0; s < 2; ++s) {       // two k=32 WMMA steps
            U16 af[2], bf_[4];
#pragma unroll
            for (int fa = 0; fa < 2; ++fa) {
                int row = wm * 32 + fa * 16 + l16;
#pragma unroll
                for (int v = 0; v < 8; ++v) {
                    int kk = (v < 4 ? 2 * v : 2 * v + 8) + 8 * half;
                    af[fa].u[v] = *(const unsigned*)&As[row][s * 32 + kk];
                }
            }
#pragma unroll
            for (int fb = 0; fb < 4; ++fb) {
                int col = wn * 64 + fb * 16 + l16;
#pragma unroll
                for (int v = 0; v < 8; ++v)
                    bf_[fb].u[v] = *(const unsigned*)&Bt[col][s * 32 + (v < 4 ? 2 * v : 2 * v + 8) + 8 * half];
            }
#pragma unroll
            for (int fa = 0; fa < 2; ++fa)
#pragma unroll
                for (int fb = 0; fb < 4; ++fb)
                    acc[fa][fb] = wmma_bf16(af[fa], bf_[fb], acc[fa][fb]);
        }
        __syncthreads();   // protect LDS against next stage's stores
    }

    // epilogue: bias, activation, residual, store
#pragma unroll
    for (int fa = 0; fa < 2; ++fa) {
#pragma unroll
        for (int fb = 0; fb < 4; ++fb) {
            int col = n0 + wn * 64 + fb * 16 + l16;
            float bv = bias[col];
#pragma unroll
            for (int r = 0; r < 8; ++r) {
                int row = m0 + wm * 32 + fa * 16 + r + 8 * half;
                float vv = acc[fa][fb][r] + bv;
                if (ACT == 1) vv = vv * fast_sigmoid(vv);   // SiLU
                if (HAS_RESID) vv += resid[(size_t)row * N + col];
                if (NT_STORE)
                    __builtin_nontemporal_store(vv, &C[(size_t)row * N + col]);
                else
                    C[(size_t)row * N + col] = vv;
            }
        }
    }
}

// ---------------------------------------------------------------------------
// 4) Sigmoid attention (causal), fused with the u-gate multiply.
//    grid = (4 qblocks, H, B), 256 threads = 8 waves; each wave owns 16 queries.
//    Streaming over 32-key blocks: S = Q Kᵀ; P = sigmoid(S*scale+rb)·mask;
//    AO += P V.  aou = AO * u.
// ---------------------------------------------------------------------------
__global__ __launch_bounds__(256)
void attn_kernel(const float* __restrict__ qkv, const float* __restrict__ u,
                 const float* __restrict__ relb, float* __restrict__ aou)
{
    const int qb = blockIdx.x, h = blockIdx.y, b = blockIdx.z;
    const int t = threadIdx.x, lane = t & 31, w = t >> 5;
    const int half = lane >> 4, l16 = lane & 15;
    const int q0 = qb * 128;
    const float scale = 0.125f;            // 1/sqrt(DH=64)
    const float rb = relb[h];

    __shared__ unsigned short Qs[128][66];     // [q][dh]
    __shared__ unsigned short Ks[32][66];      // [key][dh]  == Bt for scores
    __shared__ unsigned short Vt[64][34];      // [dh][key]  == Bt for PV
    __shared__ unsigned short Ps[8][16][34];   // per-wave P repack buffer

    // stage Q block
#pragma unroll
    for (int i = 0; i < 32; ++i) {
        int idx = i * 256 + t;
        int r = idx >> 6, c = idx & 63;
        Qs[r][c] = f2bf(qkv[(size_t)(b * Sn + q0 + r) * (3 * Dn) + h * DHn + c]);
    }
    __syncthreads();

    // Q A-fragments (K-dim 64 -> two x32 frags), held for the whole kernel
    U16 qf[2];
    {
        int row = w * 16 + l16;
#pragma unroll
        for (int s = 0; s < 2; ++s)
#pragma unroll
            for (int v = 0; v < 8; ++v) {
                int kk = (v < 4 ? 2 * v : 2 * v + 8) + 8 * half;
                qf[s].u[v] = *(const unsigned*)&Qs[row][s * 32 + kk];
            }
    }

    const v8f zero = {0.f, 0.f, 0.f, 0.f, 0.f, 0.f, 0.f, 0.f};
    v8f ao[4];
    ao[0] = zero; ao[1] = zero; ao[2] = zero; ao[3] = zero;

    const int qmaxtile = q0 + w * 16 + 15;   // last query this wave owns
    const int kend = q0 + 128;               // keys needed by this block

    for (int kb0 = 0; kb0 < kend; kb0 += 32) {
        __syncthreads();
        // stage K block [32][64] and V block transposed [64][32]
#pragma unroll
        for (int i = 0; i < 8; ++i) {
            int idx = i * 256 + t;
            int r = idx >> 6, c = idx & 63;
            Ks[r][c] = f2bf(qkv[(size_t)(b * Sn + kb0 + r) * (3 * Dn) + Dn + h * DHn + c]);
        }
#pragma unroll
        for (int i = 0; i < 8; ++i) {
            int idx = i * 256 + t;
            int r = idx >> 6, c = idx & 63;
            Vt[c][r] = f2bf(qkv[(size_t)(b * Sn + kb0 + r) * (3 * Dn) + 2 * Dn + h * DHn + c]);
        }
        __syncthreads();

        if (kb0 > qmaxtile) continue;        // fully masked for this wave

        // scores: two 16x16 tiles (keys kb0..+15, +16..+31)
#pragma unroll
        for (int j = 0; j < 2; ++j) {
            v8f sc = zero;
#pragma unroll
            for (int s = 0; s < 2; ++s) {
                U16 bf_;
#pragma unroll
                for (int v = 0; v < 8; ++v)
                    bf_.u[v] = *(const unsigned*)&Ks[j * 16 + l16][s * 32 + (v < 4 ? 2 * v : 2 * v + 8) + 8 * half];
                sc = wmma_bf16(qf[s], bf_, sc);
            }
            // sigmoid + causal mask, write P tile (bf16) to per-wave LDS
#pragma unroll
            for (int r = 0; r < 8; ++r) {
                int qg = q0 + w * 16 + r + 8 * half;
                int kg = kb0 + j * 16 + l16;
                float vsc = sc[r] * scale + rb;
                float p = (kg <= qg) ? fast_sigmoid(vsc) : 0.f;
                Ps[w][r + 8 * half][j * 16 + l16] = f2bf(p);
            }
        }

        // repack P as an A-fragment (16 q x 32 keys)
        U16 pf;
#pragma unroll
        for (int v = 0; v < 8; ++v) {
            int kk = (v < 4 ? 2 * v : 2 * v + 8) + 8 * half;
            pf.u[v] = *(const unsigned*)&Ps[w][l16][kk];
        }
        // AO += P @ V  (4 dh tiles of 16)
#pragma unroll
        for (int j2 = 0; j2 < 4; ++j2) {
            U16 vf;
#pragma unroll
            for (int v = 0; v < 8; ++v)
                vf.u[v] = *(const unsigned*)&Vt[j2 * 16 + l16][(v < 4 ? 2 * v : 2 * v + 8) + 8 * half];
            ao[j2] = wmma_bf16(pf, vf, ao[j2]);
        }
    }

    // epilogue: aou = ao * u   (head h occupies cols h*64..h*64+63 of D)
#pragma unroll
    for (int j2 = 0; j2 < 4; ++j2) {
        int col = h * DHn + j2 * 16 + l16;
#pragma unroll
        for (int r = 0; r < 8; ++r) {
            int qg = q0 + w * 16 + r + 8 * half;
            size_t idx = (size_t)(b * Sn + qg) * Dn + col;
            aou[idx] = ao[j2][r] * u[idx];
        }
    }
}

// ---------------------------------------------------------------------------
// Host-side orchestration
// ---------------------------------------------------------------------------
extern "C" void kernel_launch(void* const* d_in, const int* in_sizes, int n_in,
                              void* d_out, int out_size, void* d_ws, size_t ws_size,
                              hipStream_t stream)
{
    (void)in_sizes; (void)n_in; (void)out_size; (void)ws_size;

    const int*   ids    = (const int*)  d_in[0];
    const float* emb    = (const float*)d_in[1];
    const float* pos    = (const float*)d_in[2];
    const float* qkv_w  = (const float*)d_in[3];
    const float* qkv_b  = (const float*)d_in[4];
    const float* gate_w = (const float*)d_in[5];
    const float* gate_b = (const float*)d_in[6];
    const float* out_w  = (const float*)d_in[7];
    const float* out_b  = (const float*)d_in[8];
    const float* relb   = (const float*)d_in[9];
    const float* ln_g   = (const float*)d_in[10];
    const float* ln_b   = (const float*)d_in[11];
    const float* proj_w = (const float*)d_in[12];
    const float* proj_b = (const float*)d_in[13];
    float* logits = (float*)d_out;

    // workspace carve-out (floats): x, xn, qkv, u, aou  -> ~112 MB total
    float* ws  = (float*)d_ws;
    float* x   = ws;
    float* xn  = x   + (size_t)MROWS * Dn;
    float* qkv = xn  + (size_t)MROWS * Dn;
    float* u   = qkv + (size_t)MROWS * 3 * Dn;
    float* aou = u   + (size_t)MROWS * Dn;

    embed_kernel<<<MROWS, 256, 0, stream>>>(ids, emb, pos, x);

    for (int l = 0; l < Ln; ++l) {
        ln_kernel<<<MROWS, 256, 0, stream>>>(x, ln_g + (size_t)l * Dn, ln_b + (size_t)l * Dn, xn);

        gemm_bf16_kernel<0, false, false><<<dim3(3 * Dn / 128, MROWS / 128), 256, 0, stream>>>(
            xn, qkv_w + (size_t)l * Dn * 3 * Dn, qkv_b + (size_t)l * 3 * Dn,
            nullptr, qkv, MROWS, 3 * Dn, Dn);

        gemm_bf16_kernel<1, false, false><<<dim3(Dn / 128, MROWS / 128), 256, 0, stream>>>(
            xn, gate_w + (size_t)l * Dn * Dn, gate_b + (size_t)l * Dn,
            nullptr, u, MROWS, Dn, Dn);

        attn_kernel<<<dim3(Sn / 128, Hn, Bn), 256, 0, stream>>>(
            qkv, u, relb + (size_t)l * Hn, aou);

        gemm_bf16_kernel<0, true, false><<<dim3(Dn / 128, MROWS / 128), 256, 0, stream>>>(
            aou, out_w + (size_t)l * Dn * Dn, out_b + (size_t)l * Dn,
            /*resid=*/x, x, MROWS, Dn, Dn);
    }

    gemm_bf16_kernel<0, false, true><<<dim3(Vn / 128, MROWS / 128), 256, 0, stream>>>(
        x, proj_w, proj_b, nullptr, logits, MROWS, Vn, Dn);
}
